// Block_46926812676945
// MI455X (gfx1250) — matrix operand order repaired and checked by reading, not compiled, not run on hardware
//
#include <hip/hip_runtime.h>
#include <hip/hip_bf16.h>

typedef __attribute__((ext_vector_type(16))) _Float16 v16h;
typedef __attribute__((ext_vector_type(8)))  _Float16 v8h;
typedef __attribute__((ext_vector_type(8)))  float    v8f;
typedef __attribute__((ext_vector_type(4)))  unsigned int v4u;
typedef __attribute__((ext_vector_type(8)))  int      v8i;
typedef __attribute__((ext_vector_type(4)))  int      v4i;

#define C_EMBD 768
#define T_SEQ  1024
#define NH     12
#define HD     64
#define BB     8
#define M_ROWS (BB * T_SEQ)   /* 8192 */
#define FF_DIM 3072

static __device__ __forceinline__ v8f wmma_f16(v16h a, v16h b, v8f c) {
  return __builtin_amdgcn_wmma_f32_16x16x32_f16(false, a, false, b, (short)0, c,
                                                false, false);
}

// A-fragment (16x32 f16, MxK): lane holds row m=lane&15; element e is
// K = kbase + (lane>>4)*16 + e.
// `p` must already point at row*stride + kbase + (lane>>4)*16.
static __device__ __forceinline__ v16h make_afrag(const _Float16* p) {
  v8h lo = *(const v8h*)p;
  v8h hi = *(const v8h*)(p + 8);
  return __builtin_shufflevector(lo, hi, 0, 1, 2, 3, 4, 5, 6, 7,
                                 8, 9, 10, 11, 12, 13, 14, 15);
}

// ---------------------------------------------------------------------------
// 1-D Tensor Data Mover copy: nelem f16 elements (contiguous) global -> LDS.
// D# per CDNA5 ISA: group0 {count=1, lds_addr,
// global_addr, type=2}; group1 {data_size=1(2B), tensor_dim0=tile_dim0=nelem,
// tensor_dim1=1}. Tracked by TENSORcnt. EXEC ignored (DMA engine).
// ---------------------------------------------------------------------------
static __device__ __forceinline__ void tdm_load_1d(const void* gsrc,
                                                   unsigned lds_off,
                                                   unsigned nelem) {
  unsigned long long ga = (unsigned long long)(size_t)gsrc;
  v4u g0;
  g0.x = 1u;                                             // count=1, user mode
  g0.y = lds_off;                                        // lds_addr [63:32]
  g0.z = (unsigned)(ga & 0xFFFFFFFFu);                   // global_addr lo
  g0.w = (unsigned)((ga >> 32) & 0x01FFFFFFu) | 0x80000000u;  // hi | type=2
  v8i g1;
  g1[0] = 0x00010000;                        // data_size=1 (2 bytes)
  g1[1] = (int)((nelem & 0xFFFFu) << 16);    // tensor_dim0[15:0] @ bits 63:48
  g1[2] = (int)((nelem >> 16) & 0xFFFFu) | (1 << 16);  // dim0 hi | tensor_dim1=1
  g1[3] = (int)((nelem & 0xFFFFu) << 16);    // tile_dim0 @ bits 127:112
  g1[4] = 0;                                 // tile_dim1=0 (1-D), tile_dim2=0
  g1[5] = (int)nelem;                        // tensor_dim0_stride lo
  g1[6] = 0;
  g1[7] = 0;
  v4i gz = {0, 0, 0, 0};
#if __clang_major__ >= 23
  v8i gz8 = {0, 0, 0, 0, 0, 0, 0, 0};
  __builtin_amdgcn_tensor_load_to_lds(g0, g1, gz, gz, gz8, 0);
#else
  __builtin_amdgcn_tensor_load_to_lds(g0, g1, gz, gz, 0);
#endif
}

// ---------------------------------------------------------------------------
// Pack a row-major f32 [K][N] weight into f16 B-fragment layout:
// fragment f = kt*(N/16)+nt stores 32 lanes x 16 halves contiguously with
// element (lane,e) = W[kt*32 + (lane>>4)*16 + e][nt*16 + (lane&15)].
// ---------------------------------------------------------------------------
__global__ __launch_bounds__(256) void pack_b_kernel(
    const float* __restrict__ src, _Float16* __restrict__ dst, int K, int N) {
  int idx = blockIdx.x * 256 + threadIdx.x;
  if (idx >= K * N) return;
  int f = idx >> 9;
  int r = idx & 511;
  int lane = r >> 4;
  int e = r & 15;
  int ntiles = N >> 4;
  int kt = f / ntiles;
  int nt = f - kt * ntiles;
  int k = kt * 32 + ((lane >> 4) << 4) + e;
  int n = (nt << 4) + (lane & 15);
  dst[idx] = (_Float16)src[(size_t)k * N + n];
}

// ---------------------------------------------------------------------------
// RMSNorm: one wave32 per row of 768; f32 in, f16 out (scaled by g).
// ---------------------------------------------------------------------------
__global__ __launch_bounds__(256) void rmsnorm_kernel(
    const float* __restrict__ x, const float* __restrict__ g,
    _Float16* __restrict__ h) {
  int lane = threadIdx.x & 31;
  int row = blockIdx.x * 8 + (threadIdx.x >> 5);
  const float* xr = x + (size_t)row * C_EMBD;
  float ss = 0.f;
#pragma unroll
  for (int i = 0; i < 24; ++i) {
    float v = xr[lane + 32 * i];
    ss += v * v;
  }
#pragma unroll
  for (int m = 1; m < 32; m <<= 1) ss += __shfl_xor(ss, m, 32);
  float scale = rsqrtf(ss * (1.f / C_EMBD) + 1e-6f);
  _Float16* hr = h + (size_t)row * C_EMBD;
#pragma unroll
  for (int i = 0; i < 24; ++i) {
    int c = lane + 32 * i;
    hr[c] = (_Float16)(xr[c] * scale * g[c]);
  }
}

// ---------------------------------------------------------------------------
// WMMA GEMM with TDM-staged B operand: out[M x N] = A[M x K] * Bfrag.
// Block = 8 waves -> 32x256 tile; each wave owns a 16x64 sub-tile.
// Per K-step the block consumes one contiguous 16 KB slab of B fragments
// (16 frags x 512 halves); wave 0 DMAs slab kb+1 into the alternate LDS
// buffer via TENSOR_LOAD_TO_LDS while all waves compute from slab kb.
// EPI 0: f16 row-major | 1: f16 V-transpose [b][h][d][t]
// EPI 2: f16 gelu(acc+bias) | 3: f32 acc+bias+residual
// ---------------------------------------------------------------------------
template <int EPI>
__global__ __launch_bounds__(256) void gemm_wmma(
    const _Float16* __restrict__ A, const _Float16* __restrict__ Bf,
    void* __restrict__ outp, const float* __restrict__ bias,
    const float* __restrict__ res, int K, int N) {
  __shared__ __align__(32) _Float16 Blds[2][16 * 512];  // 2 x 16 KB
  const int lane = threadIdx.x & 31;
  const int w = threadIdx.x >> 5;
  const int half = lane >> 4;
  const int m0 = (blockIdx.x * 2 + (w >> 2)) * 16;
  const int n0 = (blockIdx.y * 4 + (w & 3)) * 64;

  v8f zero = {};
  v8f acc[4];
#pragma unroll
  for (int j = 0; j < 4; ++j) acc[j] = zero;

  const _Float16* Arow = A + (size_t)(m0 + (lane & 15)) * K + half * 16;
  const int nfr = N >> 4;
  const int kbn = K >> 5;
  // source slab for K-step kb: fragments [kb*nfr + blockIdx.y*16, +16)
  const _Float16* Bslab0 = Bf + (size_t)blockIdx.y * 16 * 512;
  const unsigned slab_bytes_step = (unsigned)nfr * 512u * 2u;

  if (w == 0) {
    tdm_load_1d(Bslab0, (unsigned)(size_t)&Blds[0][0], 16 * 512);
  }

  for (int kb = 0; kb < kbn; ++kb) {
    const int p = kb & 1;
    if (w == 0) {
      if (kb + 1 < kbn) {
        tdm_load_1d((const char*)Bslab0 + (size_t)(kb + 1) * slab_bytes_step,
                    (unsigned)(size_t)&Blds[1 - p][0], 16 * 512);
        __builtin_amdgcn_s_wait_tensorcnt(1);  // oldest (buf p) complete
      } else {
        __builtin_amdgcn_s_wait_tensorcnt(0);
      }
    }
    __syncthreads();  // publish buf p to all waves

    const _Float16* ap = Arow + kb * 32;
    __builtin_prefetch(ap + 64, 0, 0);
    v16h a = make_afrag(ap);
    const _Float16* bl = &Blds[p][((w & 3) * 4) * 512 + lane * 16];
    v16h b0 = *(const v16h*)(bl);
    v16h b1 = *(const v16h*)(bl + 512);
    v16h b2 = *(const v16h*)(bl + 1024);
    v16h b3 = *(const v16h*)(bl + 1536);
    acc[0] = wmma_f16(a, b0, acc[0]);
    acc[1] = wmma_f16(a, b1, acc[1]);
    acc[2] = wmma_f16(a, b2, acc[2]);
    acc[3] = wmma_f16(a, b3, acc[3]);

    __syncthreads();  // all reads of buf p done before it is re-filled
  }

#pragma unroll
  for (int j = 0; j < 4; ++j) {
    int n = n0 + j * 16 + (lane & 15);
    float bv = (EPI == 2 || EPI == 3) ? bias[n] : 0.f;
#pragma unroll
    for (int r = 0; r < 8; ++r) {
      int m = m0 + r + half * 8;
      float v = acc[j][r];
      if (EPI == 0) {
        ((_Float16*)outp)[(size_t)m * N + n] = (_Float16)v;
      } else if (EPI == 1) {
        int b = m >> 10, t = m & 1023;
        int hh = n >> 6, d = n & 63;
        ((_Float16*)outp)[(((size_t)b * NH + hh) * HD + d) * T_SEQ + t] =
            (_Float16)v;
      } else if (EPI == 2) {
        float u = v + bv;
        float gl = 0.5f * u *
                   (1.f + tanhf(0.7978845608028654f *
                                (u + 0.044715f * u * u * u)));
        ((_Float16*)outp)[(size_t)m * N + n] = (_Float16)gl;
      } else {
        ((float*)outp)[(size_t)m * N + n] = v + bv + res[(size_t)m * N + n];
      }
    }
  }
}

// ---------------------------------------------------------------------------
// Sliding-window (64) causal attention: one wave per (b, h, 16-query tile).
// Q,K: f16 row-major [8192][768] (col = h*64+d). Vt: f16 [b][h][d][t].
// ---------------------------------------------------------------------------
__global__ __launch_bounds__(256) void attn_kernel(
    const _Float16* __restrict__ Q, const _Float16* __restrict__ Kf,
    const _Float16* __restrict__ Vt, _Float16* __restrict__ O) {
  __shared__ __align__(32) _Float16 Plds[8][16 * 96];
  const int lane = threadIdx.x & 31;
  const int w = threadIdx.x >> 5;
  const int half = lane >> 4;
  const int wid = blockIdx.x * 8 + w;
  const int qt = wid & 63;
  const int bh = wid >> 6;
  const int h = bh % NH;
  const int b = bh / NH;
  const int t0 = qt * 16;

  v8f zero = {};
  v16h zero16 = {};
  v8f sc[5];
#pragma unroll
  for (int j = 0; j < 5; ++j) sc[j] = zero;

  const int qrow = b * T_SEQ + t0 + (lane & 15);
  const _Float16* qp = Q + (size_t)qrow * C_EMBD + h * HD + half * 16;

#pragma unroll
  for (int j = 0; j < 5; ++j) {
    int ktile = qt - 4 + j;
    if (ktile >= 0) {  // wave-uniform branch: EXEC all-ones or all-zero(skip)
      int s0 = ktile * 16;
      const int krow = b * T_SEQ + s0 + (lane & 15);
      const _Float16* kp = Kf + (size_t)krow * C_EMBD + h * HD + half * 16;
#pragma unroll
      for (int c2 = 0; c2 < 2; ++c2) {
        v16h a = make_afrag(qp + c2 * 32);
        v16h bb = *(const v16h*)(kp + c2 * 32);
        sc[j] = wmma_f16(a, bb, sc[j]);
      }
    }
  }

  // mask + softmax; query row m = r + 8*half lives within one 16-lane half
#pragma unroll
  for (int r = 0; r < 8; ++r) {
    int t = t0 + r + half * 8;
    float svv[5];
    float mx = -1e30f;
#pragma unroll
    for (int j = 0; j < 5; ++j) {
      int s = (qt - 4 + j) * 16 + (lane & 15);
      bool valid = (qt - 4 + j >= 0) && (s <= t) && (t - s < 64);
      svv[j] = valid ? sc[j][r] * 0.125f : -1e30f;
      mx = fmaxf(mx, svv[j]);
    }
#pragma unroll
    for (int msk = 1; msk < 16; msk <<= 1) mx = fmaxf(mx, __shfl_xor(mx, msk, 32));
    float sum = 0.f;
    float pe[5];
#pragma unroll
    for (int j = 0; j < 5; ++j) {
      pe[j] = __expf(svv[j] - mx);
      sum += pe[j];
    }
#pragma unroll
    for (int msk = 1; msk < 16; msk <<= 1) sum += __shfl_xor(sum, msk, 32);
    float inv = 1.f / sum;
    int mrow = r + half * 8;
#pragma unroll
    for (int j = 0; j < 5; ++j)
      Plds[w][mrow * 96 + j * 16 + (lane & 15)] = (_Float16)(pe[j] * inv);
    Plds[w][mrow * 96 + 80 + (lane & 15)] = (_Float16)0.f;  // pad cols 80..95
  }

  asm volatile("s_wait_dscnt 0x0" ::: "memory");  // wave-local LDS RAW fence

  v8f o[4];
#pragma unroll
  for (int j = 0; j < 4; ++j) o[j] = zero;
  const int s0base = (qt - 4) * 16;

#pragma unroll
  for (int c2 = 0; c2 < 3; ++c2) {
    v16h pa = make_afrag(&Plds[w][(lane & 15) * 96 + c2 * 32 + half * 16]);
    int cb = s0base + c2 * 32 + half * 16;  // 16-aligned: fully <0 or >=0
#pragma unroll
    for (int j = 0; j < 4; ++j) {
      int vrow = (b * NH + h) * HD + j * 16 + (lane & 15);
      v16h vb = zero16;
      if (cb >= 0) vb = *(const v16h*)(Vt + (size_t)vrow * T_SEQ + cb);
      o[j] = wmma_f16(pa, vb, o[j]);
    }
  }

#pragma unroll
  for (int j = 0; j < 4; ++j) {
#pragma unroll
    for (int r = 0; r < 8; ++r) {
      int m = b * T_SEQ + t0 + r + half * 8;
      int n = h * HD + j * 16 + (lane & 15);
      O[(size_t)m * C_EMBD + n] = (_Float16)o[j][r];
    }
  }
}

__global__ void aux_kernel(float* out) {
  if (threadIdx.x == 0) out[(size_t)M_ROWS * C_EMBD] = 0.f;
}

extern "C" void kernel_launch(void* const* d_in, const int* in_sizes, int n_in,
                              void* d_out, int out_size, void* d_ws,
                              size_t ws_size, hipStream_t stream) {
  (void)in_sizes; (void)n_in; (void)out_size; (void)ws_size;
  const float* x      = (const float*)d_in[0];
  const float* w_q    = (const float*)d_in[1];
  const float* w_k    = (const float*)d_in[2];
  const float* w_v    = (const float*)d_in[3];
  const float* w_proj = (const float*)d_in[4];
  const float* b_proj = (const float*)d_in[5];
  const float* w_ff1  = (const float*)d_in[6];
  const float* b_ff1  = (const float*)d_in[7];
  const float* w_ff2  = (const float*)d_in[8];
  const float* b_ff2  = (const float*)d_in[9];
  const float* g1     = (const float*)d_in[10];
  const float* g2     = (const float*)d_in[11];
  float* out = (float*)d_out;

  char* ws = (char*)d_ws;
  size_t off = 0;
  auto alloc = [&](size_t bytes) {
    char* p = ws + off;
    off = (off + bytes + 255) & ~(size_t)255;
    return p;
  };
  const size_t WD  = (size_t)C_EMBD * C_EMBD * 2;
  const size_t WF  = (size_t)C_EMBD * FF_DIM * 2;
  const size_t ACT = (size_t)M_ROWS * C_EMBD * 2;
  const size_t A32 = (size_t)M_ROWS * C_EMBD * 4;
  const size_t FFA = (size_t)M_ROWS * FF_DIM * 2;

  _Float16* WqF  = (_Float16*)alloc(WD);
  _Float16* WkF  = (_Float16*)alloc(WD);
  _Float16* WvF  = (_Float16*)alloc(WD);
  _Float16* WpF  = (_Float16*)alloc(WD);
  _Float16* Wf1F = (_Float16*)alloc(WF);
  _Float16* Wf2F = (_Float16*)alloc(WF);
  _Float16* h1   = (_Float16*)alloc(ACT);
  _Float16* Qb   = (_Float16*)alloc(ACT);
  _Float16* Kb   = (_Float16*)alloc(ACT);
  _Float16* Vt   = (_Float16*)alloc(ACT);
  _Float16* attn = (_Float16*)alloc(ACT);
  float*    x2   = (float*)   alloc(A32);
  _Float16* h2   = (_Float16*)alloc(ACT);
  _Float16* ffa  = (_Float16*)alloc(FFA);

  const int nW = C_EMBD * C_EMBD;
  const int nF = C_EMBD * FF_DIM;
  pack_b_kernel<<<(nW + 255) / 256, 256, 0, stream>>>(w_q,    WqF,  C_EMBD, C_EMBD);
  pack_b_kernel<<<(nW + 255) / 256, 256, 0, stream>>>(w_k,    WkF,  C_EMBD, C_EMBD);
  pack_b_kernel<<<(nW + 255) / 256, 256, 0, stream>>>(w_v,    WvF,  C_EMBD, C_EMBD);
  pack_b_kernel<<<(nW + 255) / 256, 256, 0, stream>>>(w_proj, WpF,  C_EMBD, C_EMBD);
  pack_b_kernel<<<(nF + 255) / 256, 256, 0, stream>>>(w_ff1,  Wf1F, C_EMBD, FF_DIM);
  pack_b_kernel<<<(nF + 255) / 256, 256, 0, stream>>>(w_ff2,  Wf2F, FF_DIM, C_EMBD);

  rmsnorm_kernel<<<M_ROWS / 8, 256, 0, stream>>>(x, g1, h1);

  dim3 g768(M_ROWS / 32, C_EMBD / 256);
  dim3 gff (M_ROWS / 32, FF_DIM / 256);
  gemm_wmma<0><<<g768, 256, 0, stream>>>(h1, WqF, Qb, nullptr, nullptr, C_EMBD, C_EMBD);
  gemm_wmma<0><<<g768, 256, 0, stream>>>(h1, WkF, Kb, nullptr, nullptr, C_EMBD, C_EMBD);
  gemm_wmma<1><<<g768, 256, 0, stream>>>(h1, WvF, Vt, nullptr, nullptr, C_EMBD, C_EMBD);

  attn_kernel<<<(BB * NH * (T_SEQ / 16)) / 8, 256, 0, stream>>>(Qb, Kb, Vt, attn);

  gemm_wmma<3><<<g768, 256, 0, stream>>>(attn, WpF, x2, b_proj, x, C_EMBD, C_EMBD);
  rmsnorm_kernel<<<M_ROWS / 8, 256, 0, stream>>>(x2, g2, h2);
  gemm_wmma<2><<<gff, 256, 0, stream>>>(h2, Wf1F, ffa, b_ff1, nullptr, C_EMBD, FF_DIM);
  gemm_wmma<3><<<g768, 256, 0, stream>>>(ffa, Wf2F, out, b_ff2, x2, FF_DIM, C_EMBD);
  aux_kernel<<<1, 32, 0, stream>>>(out);
}
